// MultiResBiplane_41635412968145
// MI455X (gfx1250) — matrix operand order repaired and checked by reading, not compiled, not run on hardware
//
#include <hip/hip_runtime.h>
#include <hip/hip_bf16.h>
#include <math.h>

typedef __attribute__((ext_vector_type(16))) _Float16 v16h;
typedef __attribute__((ext_vector_type(8)))  _Float16 v8h;
typedef __attribute__((ext_vector_type(8)))  float    v8f;

#define FEAT        16
#define IN_DIM      72
#define HID         36
#define K1P         96   // IN_DIM padded to 3x32
#define N1P         48   // HID padded to 3x16
#define K2P         64   // HID padded to 2x32
#define N2P         16
#define NPTS_TOTAL  (8*4096*32)
#define BLOCK       256
#define WAVES       8

// ---------------- plane transpose: [C,H,W] f32 -> [H*W, C] f16 ----------------
__global__ __launch_bounds__(256) void transpose_plane_kernel(
    const float* __restrict__ plane, _Float16* __restrict__ out, int HW) {
  int t = blockIdx.x * blockDim.x + threadIdx.x;
  if (t >= HW) return;
  v8h h0, h1;
#pragma unroll
  for (int c = 0; c < 8; ++c) h0[c] = (_Float16)plane[c * HW + t];        // coalesced per c
#pragma unroll
  for (int c = 0; c < 8; ++c) h1[c] = (_Float16)plane[(8 + c) * HW + t];
  v8h* o = (v8h*)(out + (size_t)t * 16);                                  // coalesced 32B/thread
  o[0] = h0; o[1] = h1;
}

// ---------------- fragment loaders (CDNA5 16-bit WMMA layouts) ----------------
// A 16x32 f16: lane half h, element i holds K = h*16 + i (contiguous, matches B)
__device__ __forceinline__ v16h load_afrag(const _Float16* __restrict__ row, int s, int half) {
  v16h a;
#pragma unroll
  for (int i = 0; i < 4; ++i) {
    int kA = s * 32 + half * 16 + 2 * i;
    int kB = kA + 8;
    a[2*i]     = row[kA];
    a[2*i + 1] = row[kA + 1];
    a[8 + 2*i] = row[kB];
    a[9 + 2*i] = row[kB + 1];
  }
  return a;
}
// B 32x16 f16: lane = N (mod 16), per-lane K sequential: K = half*16 + i  -> contiguous 32B
__device__ __forceinline__ v16h load_bfrag(const _Float16* __restrict__ p) {
  v8h b0 = *(const v8h*)p;
  v8h b1 = *(const v8h*)(p + 8);
  v16h b;
#pragma unroll
  for (int i = 0; i < 8; ++i) { b[i] = b0[i]; b[8 + i] = b1[i]; }
  return b;
}

// ---------------- bilinear sample one plane -> 16 f16 into LDS row ----------------
__device__ __forceinline__ void sample_plane(const _Float16* __restrict__ pl, int R,
                                             float xf, float yf, _Float16* __restrict__ dst) {
  float fx = (xf + 1.f) * 0.5f * (float)(R - 1);
  float fy = (yf + 1.f) * 0.5f * (float)(R - 1);
  float x0f = floorf(fx), y0f = floorf(fy);
  float wx1 = fx - x0f, wx0 = 1.f - wx1;
  float wy1 = fy - y0f, wy0 = 1.f - wy1;
  int x0 = (int)x0f, y0 = (int)y0f;
  int x1 = x0 + 1, y1 = y0 + 1;
  float vx0 = (x0 >= 0 && x0 < R) ? 1.f : 0.f;
  float vx1 = (x1 >= 0 && x1 < R) ? 1.f : 0.f;
  float vy0 = (y0 >= 0 && y0 < R) ? 1.f : 0.f;
  float vy1 = (y1 >= 0 && y1 < R) ? 1.f : 0.f;
  int x0c = min(max(x0, 0), R - 1), x1c = min(max(x1, 0), R - 1);
  int y0c = min(max(y0, 0), R - 1), y1c = min(max(y1, 0), R - 1);
  float w00 = wy0 * wx0 * vy0 * vx0, w01 = wy0 * wx1 * vy0 * vx1;
  float w10 = wy1 * wx0 * vy1 * vx0, w11 = wy1 * wx1 * vy1 * vx1;
  float acc[16];
#pragma unroll
  for (int c = 0; c < 16; ++c) acc[c] = 0.f;
  const int i00 = (y0c * R + x0c) * 16, i01 = (y0c * R + x1c) * 16;
  const int i10 = (y1c * R + x0c) * 16, i11 = (y1c * R + x1c) * 16;
  {
    const v8h* q = (const v8h*)(pl + i00); v8h u0 = q[0], u1 = q[1];
#pragma unroll
    for (int c = 0; c < 8; ++c) { acc[c] += w00 * (float)u0[c]; acc[8 + c] += w00 * (float)u1[c]; }
  }
  {
    const v8h* q = (const v8h*)(pl + i01); v8h u0 = q[0], u1 = q[1];
#pragma unroll
    for (int c = 0; c < 8; ++c) { acc[c] += w01 * (float)u0[c]; acc[8 + c] += w01 * (float)u1[c]; }
  }
  {
    const v8h* q = (const v8h*)(pl + i10); v8h u0 = q[0], u1 = q[1];
#pragma unroll
    for (int c = 0; c < 8; ++c) { acc[c] += w10 * (float)u0[c]; acc[8 + c] += w10 * (float)u1[c]; }
  }
  {
    const v8h* q = (const v8h*)(pl + i11); v8h u0 = q[0], u1 = q[1];
#pragma unroll
    for (int c = 0; c < 8; ++c) { acc[c] += w11 * (float)u0[c]; acc[8 + c] += w11 * (float)u1[c]; }
  }
  v8h r0, r1;
#pragma unroll
  for (int c = 0; c < 8; ++c) { r0[c] = (_Float16)acc[c]; r1[c] = (_Float16)acc[8 + c]; }
  *(v8h*)dst = r0;
  *(v8h*)(dst + 8) = r1;
}

// ---------------- fused sample + MLP (WMMA) ----------------
__global__ __launch_bounds__(BLOCK) void fused_biplane_mlp_kernel(
    const float* __restrict__ coords, const float* __restrict__ tfeat,
    const float* __restrict__ w1, const float* __restrict__ b1v,
    const float* __restrict__ w2, const float* __restrict__ b2v,
    const _Float16* __restrict__ p0, const _Float16* __restrict__ p1,
    const _Float16* __restrict__ p2, float* __restrict__ out) {
  __shared__ _Float16 sW1[N1P * K1P];             // [n][k], zero padded
  __shared__ _Float16 sW2[N2P * K2P];             // [n][k], zero padded
  __shared__ float    sB1[N1P];
  __shared__ float    sB2[2];
  __shared__ __align__(16) _Float16 sTh[24];
  __shared__ __align__(16) _Float16 sAH[WAVES][32][K1P];  // A tile; hidden acts alias rows
  __shared__ float    sOut[WAVES][32][2];

  const int tid  = threadIdx.x;
  const int wv   = tid >> 5;
  const int lane = tid & 31;
  const int lm   = lane & 15;
  const int half = lane >> 4;
  const int img  = blockIdx.x >> 9;               // 512 blocks of 256 pts per image

  // ---- stage weights / biases / t-features ----
  for (int idx = tid; idx < N1P * K1P; idx += BLOCK) {
    int n = idx / K1P, k = idx - n * K1P;
    sW1[idx] = (n < HID && k < IN_DIM) ? (_Float16)w1[n * IN_DIM + k] : (_Float16)0.f;
  }
  for (int idx = tid; idx < N2P * K2P; idx += BLOCK) {
    int n = idx >> 6, k = idx & 63;
    sW2[idx] = (n < 2 && k < HID) ? (_Float16)w2[n * HID + k] : (_Float16)0.f;
  }
  if (tid < N1P) sB1[tid] = (tid < HID) ? b1v[tid] : 0.f;
  if (tid < 2)   sB2[tid] = b2v[tid];
  if (tid < 24)  sTh[tid] = (_Float16)tfeat[img * 24 + tid];
  __syncthreads();

  // ---- per-point sampling into this wave's A tile (row = lane) ----
  const int g = blockIdx.x * BLOCK + tid;
  const float xf = coords[2 * g], yf = coords[2 * g + 1];
  _Float16* myrow = &sAH[wv][lane][0];
  sample_plane(p0, 128, xf, yf, myrow + 0);
  sample_plane(p1, 256, xf, yf, myrow + 16);
  sample_plane(p2, 512, xf, yf, myrow + 32);
  *(v8h*)(myrow + 48) = *(const v8h*)&sTh[0];     // t-feats, cols 48..71
  *(v8h*)(myrow + 56) = *(const v8h*)&sTh[8];
  *(v8h*)(myrow + 64) = *(const v8h*)&sTh[16];
  v8h z;
#pragma unroll
  for (int i = 0; i < 8; ++i) z[i] = (_Float16)0.f;
  *(v8h*)(myrow + 72) = z;                        // K padding must be NaN-free
  *(v8h*)(myrow + 80) = z;
  *(v8h*)(myrow + 88) = z;
  asm volatile("s_wait_dscnt 0x0" ::: "memory");  // wave-local LDS RAW fence

  // ---- layer 1: [16x96] x [96x48] per M-tile, f32 accumulate ----
#pragma unroll
  for (int mt = 0; mt < 2; ++mt) {
    const _Float16* arow = &sAH[wv][mt * 16 + lm][0];
    v16h a0 = load_afrag(arow, 0, half);
    v16h a1 = load_afrag(arow, 1, half);
    v16h a2 = load_afrag(arow, 2, half);
#pragma unroll
    for (int nt = 0; nt < 3; ++nt) {
      float bias = sB1[nt * 16 + lm];
      v8f c;
#pragma unroll
      for (int e = 0; e < 8; ++e) c[e] = bias;
      const _Float16* brow = &sW1[(nt * 16 + lm) * K1P];
      v16h b0 = load_bfrag(brow + 0 * 32 + half * 16);
      c = __builtin_amdgcn_wmma_f32_16x16x32_f16(false, a0, false, b0, (short)0, c, false, false);
      v16h bb1 = load_bfrag(brow + 1 * 32 + half * 16);
      c = __builtin_amdgcn_wmma_f32_16x16x32_f16(false, a1, false, bb1, (short)0, c, false, false);
      v16h bb2 = load_bfrag(brow + 2 * 32 + half * 16);
      c = __builtin_amdgcn_wmma_f32_16x16x32_f16(false, a2, false, bb2, (short)0, c, false, false);
      // leaky relu, write hidden acts back into tile rows (cols 0..47)
#pragma unroll
      for (int e = 0; e < 8; ++e) {
        float h = c[e];
        h = (h >= 0.f) ? h : 0.01f * h;
        sAH[wv][mt * 16 + e + 8 * half][nt * 16 + lm] = (_Float16)h;
      }
    }
  }
  asm volatile("s_wait_dscnt 0x0" ::: "memory");

  // ---- layer 2: [16x64] x [64x16] (only N<2 valid), sigmoid ----
#pragma unroll
  for (int mt = 0; mt < 2; ++mt) {
    const _Float16* hrow = &sAH[wv][mt * 16 + lm][0];
    v16h a0 = load_afrag(hrow, 0, half);
    v16h a1 = load_afrag(hrow, 1, half);
    float bias = (lm < 2) ? sB2[lm] : 0.f;
    v8f c;
#pragma unroll
    for (int e = 0; e < 8; ++e) c[e] = bias;
    const _Float16* brow = &sW2[lm * K2P];
    v16h b0 = load_bfrag(brow + 0 + half * 16);
    c = __builtin_amdgcn_wmma_f32_16x16x32_f16(false, a0, false, b0, (short)0, c, false, false);
    v16h b1f = load_bfrag(brow + 32 + half * 16);
    c = __builtin_amdgcn_wmma_f32_16x16x32_f16(false, a1, false, b1f, (short)0, c, false, false);
    if (lm < 2) {
#pragma unroll
      for (int e = 0; e < 8; ++e) {
        float v = c[e];
        v = 1.f / (1.f + __expf(-v));
        sOut[wv][mt * 16 + e + 8 * half][lm] = v;
      }
    }
  }
  asm volatile("s_wait_dscnt 0x0" ::: "memory");

  float2 o = *(float2*)&sOut[wv][lane][0];
  ((float2*)out)[g] = o;                          // coalesced 8B/lane
}

extern "C" void kernel_launch(void* const* d_in, const int* in_sizes, int n_in,
                              void* d_out, int out_size, void* d_ws, size_t ws_size,
                              hipStream_t stream) {
  const float* coords = (const float*)d_in[0];
  const float* tfeat  = (const float*)d_in[1];
  const float* plane0 = (const float*)d_in[2];
  const float* plane1 = (const float*)d_in[3];
  const float* plane2 = (const float*)d_in[4];
  const float* w1     = (const float*)d_in[5];
  const float* b1     = (const float*)d_in[6];
  const float* w2     = (const float*)d_in[7];
  const float* b2     = (const float*)d_in[8];
  float* out = (float*)d_out;

  _Float16* ws = (_Float16*)d_ws;
  _Float16* t0 = ws;                               // 128*128*16 f16
  _Float16* t1 = t0 + 128 * 128 * 16;              // 256*256*16 f16
  _Float16* t2 = t1 + 256 * 256 * 16;              // 512*512*16 f16

  transpose_plane_kernel<<<(128 * 128) / 256, 256, 0, stream>>>(plane0, t0, 128 * 128);
  transpose_plane_kernel<<<(256 * 256) / 256, 256, 0, stream>>>(plane1, t1, 256 * 256);
  transpose_plane_kernel<<<(512 * 512) / 256, 256, 0, stream>>>(plane2, t2, 512 * 512);

  fused_biplane_mlp_kernel<<<NPTS_TOTAL / BLOCK, BLOCK, 0, stream>>>(
      coords, tfeat, w1, b1, w2, b2, t0, t1, t2, out);
}
